// Model_39814346833983
// MI455X (gfx1250) — matrix operand order repaired and checked by reading, not compiled, not run on hardware
//
#include <hip/hip_runtime.h>
#include <hip/hip_bf16.h>

#define T_LEN 512
#define NBATCH 1000
#define HID 64

typedef _Float16 v16h __attribute__((ext_vector_type(16)));
typedef _Float16 v8h  __attribute__((ext_vector_type(8)));
typedef float    v8f  __attribute__((ext_vector_type(8)));

// gfx1250 has a hardware tanh transcendental (v_tanh_f32).
#if defined(__has_builtin)
#  if __has_builtin(__builtin_amdgcn_tanhf)
#    define FAST_TANH(x) __builtin_amdgcn_tanhf(x)
#  endif
#endif
#ifndef FAST_TANH
#  define FAST_TANH(x) tanhf(x)
#endif

__device__ __forceinline__ float fast_tanh(float x) { return FAST_TANH(x); }
// sigmoid(x) = 0.5*tanh(0.5x) + 0.5  (1 trans + 2 VALU, branch-free)
__device__ __forceinline__ float fast_sigm(float x) {
    return fmaf(fast_tanh(0.5f * x), 0.5f, 0.5f);
}

// ---------------------------------------------------------------------------
// Kernel 1: per-event embedding + linear + scatter into padded feat (f16).
// One block (64 threads) per event; thread j = output channel j.
// The 5 sources partition all B*T slots, so feat is fully written.
// ---------------------------------------------------------------------------
__global__ void __launch_bounds__(64)
embed_scatter_kernel(const float* __restrict__ num, int n_num,
                     const int* __restrict__ cat, int n_cat,
                     const float* __restrict__ tab,
                     const float* __restrict__ W, const float* __restrict__ bvec,
                     const int* __restrict__ bi, const int* __restrict__ ti,
                     _Float16* __restrict__ feat16, int nEvents)
{
    __shared__ float x[80];                     // in_dim <= 66
    const int e = blockIdx.x;
    if (e >= nEvents) return;
    const int j = threadIdx.x;                  // output channel 0..63
    const int ncx = n_cat * 8;
    const int in_dim = ncx + n_num;

    for (int idx = j; idx < ncx; idx += 64) {
        int c = idx >> 3, q = idx & 7;
        x[idx] = tab[cat[e * n_cat + c] * 8 + q];
    }
    for (int idx = j; idx < n_num; idx += 64) x[ncx + idx] = num[e * n_num + idx];
    __syncthreads();

    float acc = bvec[j];
    for (int i = 0; i < in_dim; ++i) acc = fmaf(x[i], W[i * HID + j], acc);

    const long token = (long)bi[e] * T_LEN + ti[e];
    feat16[token * HID + j] = (_Float16)acc;
}

// ---------------------------------------------------------------------------
// Kernel 2: layer-0 LSTM (one direction), WMMA gate GEMM fused with scalar-h
// recurrence (proj_size = 1). Block = 128 threads (4 waves) = 16 batches.
// Wave w owns gate slice [64w, 64w+64): 4 N-tiles x 2 K-halves = 8 WMMAs/step.
// All loop-invariant weights (Whh, bias, Whr, B-fragments) live in registers.
// ---------------------------------------------------------------------------
__global__ void __launch_bounds__(128)
lstm0_kernel(const _Float16* __restrict__ feat16,      // [B*T, 64]
             const float* __restrict__ Wih,            // [2, 256, 64]
             const float* __restrict__ Whh,            // [2, 256, 1]
             const float* __restrict__ Whr,            // [2, 1, 64]
             const float* __restrict__ bih,            // [2, 256]
             const float* __restrict__ bhh,            // [2, 256]
             float* __restrict__ y0)                   // [B*T, 2]
{
    const int dir  = blockIdx.y;
    const int tile = blockIdx.x;          // 63 tiles of 16 batches
    const int tid  = threadIdx.x;
    const int wave = tid >> 5;
    const int lane = tid & 31;

    __shared__ __align__(16) _Float16 wihs[256 * 64];  // 32 KB, f16 weights
    __shared__ float gates[16 * 256];                  // activated gate values
    __shared__ float h_s[16];
    __shared__ float hpart[16 * 8];

    const float* WihD = Wih + (size_t)dir * 256 * 64;
    for (int i = tid; i < 256 * 64; i += 128) wihs[i] = (_Float16)WihD[i];
    if (tid < 16) h_s[tid] = 0.0f;
    __syncthreads();

    union Frag { v16h v; v8h h[2]; };

    // Preload B fragments (Wih^T tiles) into registers from LDS.
    // B (K=32 x N=16), 16-bit: lanes 0-15 hold K=0..7 then K=16..23, lanes 16-31
    // hold K=8..15 then K=24..31, column N = lane&15.
    Frag Bf[4][2];
    {
        const int col0 = wave * 64 + (lane & 15);
        const int ks   = (lane >= 16) ? 8 : 0;
        #pragma unroll
        for (int nt = 0; nt < 4; ++nt) {
            const int col = col0 + nt * 16;
            #pragma unroll
            for (int kh = 0; kh < 2; ++kh) {
                const _Float16* p = &wihs[col * 64 + kh * 32 + ks];
                Bf[nt][kh].h[0] = *(const v8h*)(p);
                Bf[nt][kh].h[1] = *(const v8h*)(p + 16);
            }
        }
    }

    // Loop-invariant per-lane gate constants (gate phase): n = w*64+nt*16+ncol.
    const int ncol  = lane & 15;
    const int mbase = (lane >= 16) ? 8 : 0;
    float whh_r[4], bias_r[4];
    #pragma unroll
    for (int nt = 0; nt < 4; ++nt) {
        const int n = wave * 64 + nt * 16 + ncol;
        whh_r[nt]  = Whh[dir * 256 + n];
        bias_r[nt] = bih[dir * 256 + n] + bhh[dir * 256 + n];
    }
    // Activation form: out = fma(tanh(act_a * v), act_a, act_b)
    //  wave 2 (g-gate): tanh  -> a=1,   b=0
    //  waves 0/1/3    : sigm  -> a=0.5, b=0.5
    const float act_a = (wave == 2) ? 1.0f : 0.5f;
    const float act_b = (wave == 2) ? 0.0f : 0.5f;

    // Loop-invariant per-thread elementwise constants: thread covers
    // batch bl = tid>>3 and channels ch = (tid&7)*8 + j.
    const int bl = tid >> 3, kk = tid & 7;
    float whr_r[8];
    #pragma unroll
    for (int j = 0; j < 8; ++j) whr_r[j] = Whr[dir * 64 + kk * 8 + j];

    // Per-thread cell state.
    float c_reg[8];
    #pragma unroll
    for (int j = 0; j < 8; ++j) c_reg[j] = 0.0f;

    const int bg0 = tile * 16;
    // A-fragment row for this lane (16-bit A 16x32 layout):
    // lanes 0-15: K=0..7 then K=16..23 ; lanes 16-31: K=8..15 then K=24..31.
    int bg_a = bg0 + (lane & 15); if (bg_a > NBATCH - 1) bg_a = NBATCH - 1;
    const int koffA = (lane >= 16) ? 8 : 0;
    const _Float16* arow_base = feat16 + (size_t)bg_a * T_LEN * HID;

    for (int t = 0; t < T_LEN; ++t) {
        const int t_act = dir ? (T_LEN - 1 - t) : t;
        const _Float16* arow = arow_base + (size_t)t_act * HID;

        Frag Af[2];
        Af[0].h[0] = *(const v8h*)(arow + koffA);
        Af[0].h[1] = *(const v8h*)(arow + koffA + 16);
        Af[1].h[0] = *(const v8h*)(arow + 32 + koffA);
        Af[1].h[1] = *(const v8h*)(arow + 32 + koffA + 16);

        // Prefetch next timestep's row (global_prefetch_b8).
        if (t + 1 < T_LEN) {
            const int t_nxt = dir ? (t_act - 1) : (t_act + 1);
            __builtin_prefetch(arow_base + (size_t)t_nxt * HID, 0, 1);
        }

        // h_prev for the 8 batch rows this lane's C tiles cover (reused x4).
        float hprev[8];
        #pragma unroll
        for (int r = 0; r < 8; ++r) hprev[r] = h_s[mbase + r];

        #pragma unroll
        for (int nt = 0; nt < 4; ++nt) {
            v8f acc = {};
            acc = __builtin_amdgcn_wmma_f32_16x16x32_f16(
                false, Af[0].v, false, Bf[nt][0].v, (short)0, acc, false, false);
            acc = __builtin_amdgcn_wmma_f32_16x16x32_f16(
                false, Af[1].v, false, Bf[nt][1].v, (short)0, acc, false, false);

            const int n    = wave * 64 + nt * 16 + ncol;  // global gate index
            const float wh = whh_r[nt];
            const float bs = bias_r[nt];
            #pragma unroll
            for (int r = 0; r < 8; ++r) {
                const float v = fmaf(hprev[r], wh, acc[r] + bs);
                gates[(mbase + r) * 256 + n] = fmaf(fast_tanh(act_a * v), act_a, act_b);
            }
        }
        __syncthreads();

        // Elementwise c/h update: thread -> (batch bl, 8 channels).
        {
            float contrib = 0.0f;
            #pragma unroll
            for (int j = 0; j < 8; ++j) {
                const int ch = kk * 8 + j;
                const float ig = gates[bl * 256 + ch];
                const float fg = gates[bl * 256 + 64 + ch];
                const float gg = gates[bl * 256 + 128 + ch];
                const float og = gates[bl * 256 + 192 + ch];
                const float c  = fmaf(fg, c_reg[j], ig * gg);
                c_reg[j] = c;
                contrib = fmaf(og * fast_tanh(c), whr_r[j], contrib);
            }
            hpart[bl * 8 + kk] = contrib;
        }
        __syncthreads();

        if (tid < 16) {
            float h = 0.0f;
            #pragma unroll
            for (int k = 0; k < 8; ++k) h += hpart[tid * 8 + k];
            h_s[tid] = h;
            const int bgo = bg0 + tid;
            if (bgo < NBATCH) y0[((size_t)bgo * T_LEN + t_act) * 2 + dir] = h;
        }
        __syncthreads();
    }
}

// ---------------------------------------------------------------------------
// Kernel 3: layer-1 LSTM (input width 2, proj_size 1). One block (64 threads)
// per (batch, direction); thread = channel; wave shuffles + LDS for h-reduce.
// ---------------------------------------------------------------------------
__global__ void __launch_bounds__(64)
lstm1_kernel(const float* __restrict__ y0,     // [B*T, 2]
             const float* __restrict__ Wih,    // [2, 256, 2]
             const float* __restrict__ Whh,    // [2, 256, 1]
             const float* __restrict__ Whr,    // [2, 1, 64]
             const float* __restrict__ bih,    // [2, 256]
             const float* __restrict__ bhh,    // [2, 256]
             float* __restrict__ y1)           // [2, B*T]  (dir-major)
{
    const int b   = blockIdx.x;
    const int dir = blockIdx.y;
    const int ch  = threadIdx.x;                // 0..63
    __shared__ float wsum[2];

    float w0[4], w1[4], wh[4], bs[4];
    #pragma unroll
    for (int g = 0; g < 4; ++g) {
        const int n = g * 64 + ch;              // gate order i,f,g,o
        w0[g] = Wih[((size_t)dir * 256 + n) * 2 + 0];
        w1[g] = Wih[((size_t)dir * 256 + n) * 2 + 1];
        wh[g] = Whh[dir * 256 + n];
        bs[g] = bih[dir * 256 + n] + bhh[dir * 256 + n];
    }
    const float wr = Whr[dir * 64 + ch];

    float c = 0.0f, h = 0.0f;
    for (int t = 0; t < T_LEN; ++t) {
        const int t_act = dir ? (T_LEN - 1 - t) : t;
        const float x0 = y0[((size_t)b * T_LEN + t_act) * 2 + 0];
        const float x1 = y0[((size_t)b * T_LEN + t_act) * 2 + 1];

        const float gi = fast_sigm(x0 * w0[0] + x1 * w1[0] + h * wh[0] + bs[0]);
        const float gf = fast_sigm(x0 * w0[1] + x1 * w1[1] + h * wh[1] + bs[1]);
        const float gg = fast_tanh(x0 * w0[2] + x1 * w1[2] + h * wh[2] + bs[2]);
        const float go = fast_sigm(x0 * w0[3] + x1 * w1[3] + h * wh[3] + bs[3]);
        c = fmaf(gf, c, gi * gg);
        float contrib = go * fast_tanh(c) * wr;

        // wave32 reduction then combine 2 waves via LDS
        #pragma unroll
        for (int off = 16; off > 0; off >>= 1) contrib += __shfl_xor(contrib, off, 32);
        if ((threadIdx.x & 31) == 0) wsum[threadIdx.x >> 5] = contrib;
        __syncthreads();
        h = wsum[0] + wsum[1];
        if (threadIdx.x == 0)
            y1[(size_t)dir * NBATCH * T_LEN + (size_t)b * T_LEN + t_act] = h;
        __syncthreads();
    }
}

// ---------------------------------------------------------------------------
// Kernel 4: out[b,t] = mean over the 2 directions of layer-1 output.
// ---------------------------------------------------------------------------
__global__ void final_mean_kernel(const float* __restrict__ y1, float* __restrict__ out, int n)
{
    const int i = blockIdx.x * blockDim.x + threadIdx.x;
    if (i < n) out[i] = 0.5f * (y1[i] + y1[n + i]);
}

// ---------------------------------------------------------------------------
extern "C" void kernel_launch(void* const* d_in, const int* in_sizes, int n_in,
                              void* d_out, int out_size, void* d_ws, size_t ws_size,
                              hipStream_t stream)
{
    (void)n_in; (void)out_size; (void)ws_size;

    // Workspace layout
    char* ws = (char*)d_ws;
    size_t off = 0;
    _Float16* feat16 = (_Float16*)(ws + off);
    off += (size_t)NBATCH * T_LEN * HID * sizeof(_Float16);   // 65.5 MB
    off = (off + 255) & ~(size_t)255;
    float* y0 = (float*)(ws + off);
    off += (size_t)NBATCH * T_LEN * 2 * sizeof(float);        // 4.1 MB
    off = (off + 255) & ~(size_t)255;
    float* y1 = (float*)(ws + off);
    off += (size_t)NBATCH * T_LEN * 2 * sizeof(float);        // 4.1 MB

    // --- Stage 1: 5 sources -> feat16 (partition of all B*T slots) ---
    struct Src { int num, cat, tab, bi, ti, W, b, n_num, n_cat; };
    const Src srcs[5] = {
        {  0, -1, -1,  1,  2,  3,  4, 8, 0 },   // ccba
        {  5,  6,  7,  8,  9, 10, 11, 1, 2 },   // cdtx
        { 12, 13, 14, 15, 16, 17, 18, 1, 3 },   // cust
        { 19, 20, 21, 22, 23, 24, 25, 2, 8 },   // dp
        { 26, 27, 28, 29, 30, 31, 32, 1, 1 },   // remit
    };
    for (int s = 0; s < 5; ++s) {
        const Src& S = srcs[s];
        const int nEvents = in_sizes[S.bi];
        embed_scatter_kernel<<<nEvents, 64, 0, stream>>>(
            (const float*)d_in[S.num], S.n_num,
            S.cat >= 0 ? (const int*)d_in[S.cat] : nullptr, S.n_cat,
            S.tab >= 0 ? (const float*)d_in[S.tab] : nullptr,
            (const float*)d_in[S.W], (const float*)d_in[S.b],
            (const int*)d_in[S.bi], (const int*)d_in[S.ti],
            feat16, nEvents);
    }

    // --- Stage 2: layer-0 biLSTM with WMMA gate GEMM ---
    {
        dim3 grid((NBATCH + 15) / 16, 2);
        lstm0_kernel<<<grid, 128, 0, stream>>>(
            feat16,
            (const float*)d_in[33], (const float*)d_in[34], (const float*)d_in[35],
            (const float*)d_in[36], (const float*)d_in[37], y0);
    }

    // --- Stage 3: layer-1 biLSTM (input width 2) ---
    {
        dim3 grid(NBATCH, 2);
        lstm1_kernel<<<grid, 64, 0, stream>>>(
            y0,
            (const float*)d_in[38], (const float*)d_in[39], (const float*)d_in[40],
            (const float*)d_in[41], (const float*)d_in[42], y1);
    }

    // --- Stage 4: mean over directions -> d_out [B*T] ---
    {
        const int n = NBATCH * T_LEN;
        final_mean_kernel<<<(n + 255) / 256, 256, 0, stream>>>(y1, (float*)d_out, n);
    }
}
